// AttentionHead_33440615366761
// MI455X (gfx1250) — matrix operand order repaired and checked by reading, not compiled, not run on hardware
//
#include <hip/hip_runtime.h>

typedef __attribute__((ext_vector_type(16))) __bf16 v16bf;
typedef __attribute__((ext_vector_type(8)))  __bf16 v8bf;
typedef __attribute__((ext_vector_type(8)))  float  v8f;
typedef __attribute__((ext_vector_type(4)))  float  v4f;

#define B_   4
#define S_   4096
#define DIN  1024
#define DH   128
#define MTOT (B_ * S_)          // 16384 rows for the projection GEMM
#define KT   64                 // key-tile width in the flash loop

__device__ __forceinline__ v16bf mk16(v8bf lo, v8bf hi8) {
    v16bf r;
    #pragma unroll
    for (int i = 0; i < 8; ++i) { r[i] = lo[i]; r[i + 8] = hi8[i]; }
    return r;
}

__device__ __forceinline__ v16bf cvt16(v4f a, v4f b, v4f c, v4f d) {
    v16bf r;
    #pragma unroll
    for (int i = 0; i < 4; ++i) {
        r[i]      = (__bf16)a[i];
        r[i + 4]  = (__bf16)b[i];
        r[i + 8]  = (__bf16)c[i];
        r[i + 12] = (__bf16)d[i];
    }
    return r;
}

// Low 32 bits of a generic pointer to a __shared__ object = LDS byte offset
// (flat->LDS aperture mapping truncates to addr[31:0]).
__device__ __forceinline__ unsigned lds_off(const void* p) {
    return (unsigned)(size_t)p;
}

// CDNA5 async copy: 16 bytes per lane, global -> LDS, tracked by ASYNCcnt.
__device__ __forceinline__ void async_copy_b128(unsigned lds_byte, const void* gptr) {
    asm volatile("global_load_async_to_lds_b128 %0, %1, off"
                 :: "v"(lds_byte), "v"(gptr) : "memory");
}

__device__ __forceinline__ void wait_async0() {
    asm volatile("s_wait_asynccnt 0" ::: "memory");
}

// ---------------------------------------------------------------------------
// Kernel 1: fused QKV projection.  X[16384,1024] (f32) x W[1024,128] (f32)
// + bias -> bf16 in workspace.  blockIdx.y selects q/k/v.
// q stored PRE-SCALED by 1/sqrt(128); q,k row-major [B*S,128]; v stored
// TRANSPOSED per batch [B,128,S] so the attention kernel can async-copy
// V^T tiles without transposing.
// ---------------------------------------------------------------------------
__global__ __launch_bounds__(256)
void qkv_proj_kernel(const float* __restrict__ query,
                     const float* __restrict__ key,
                     const float* __restrict__ value,
                     const float* __restrict__ Wq, const float* __restrict__ bq,
                     const float* __restrict__ Wk, const float* __restrict__ bk,
                     const float* __restrict__ Wv, const float* __restrict__ bv,
                     unsigned short* __restrict__ ws_qkv) {
    __shared__ __bf16 lds_w[128 * 32];     // W tile, transposed: [n][k], 8 KB

    const int which = blockIdx.y;
    const float* X    = (which == 0) ? query : (which == 1) ? key : value;
    const float* W    = (which == 0) ? Wq    : (which == 1) ? Wk  : Wv;
    const float* bias = (which == 0) ? bq    : (which == 1) ? bk  : bv;
    const float sc    = (which == 0) ? 0.08838834764831845f : 1.0f;  // 1/sqrt(Dq)
    __bf16* outb = reinterpret_cast<__bf16*>(ws_qkv) + (size_t)which * MTOT * DH;

    const int tid  = threadIdx.x;
    const int wv   = tid >> 5;         // wave 0..7
    const int lane = tid & 31;
    const int hi   = lane >> 4;        // half-wave select
    const int lm   = lane & 15;
    const int m0   = blockIdx.x * 128 + wv * 16;   // wave's 16-row strip

    v8f acc[8] = {};

    for (int kk = 0; kk < DIN; kk += 32) {
        // --- stage W[kk..kk+32)[0..128) transposed into LDS as bf16 ---
        for (int i = tid; i < 32 * 128; i += 256) {
            int k = i >> 7, n = i & 127;
            lds_w[n * 32 + k] = (__bf16)W[(size_t)(kk + k) * DH + n];
        }
        __syncthreads();

        // --- A fragment: 16x32 bf16, rows m0+lm, ISA lane/K layout ---
        const float* xr = X + (size_t)(m0 + lm) * DIN + kk + 8 * hi;
        v4f a0 = *(const v4f*)(xr);
        v4f a1 = *(const v4f*)(xr + 4);
        v4f a2 = *(const v4f*)(xr + 16);
        v4f a3 = *(const v4f*)(xr + 20);
        v16bf afrag = cvt16(a0, a1, a2, a3);

        // --- 8 WMMAs across the 128-wide N dimension ---
        #pragma unroll
        for (int nt = 0; nt < 8; ++nt) {
            const v8bf* wr = (const v8bf*)(lds_w + (nt * 16 + lm) * 32 + 8 * hi);
            v16bf bfrag = mk16(wr[0], wr[2]);
            acc[nt] = __builtin_amdgcn_wmma_f32_16x16x32_bf16(
                false, afrag, false, bfrag, (short)0, acc[nt], false, false);
        }
        __syncthreads();
    }

    // --- bias (+scale for Q) + bf16 store ---
    if (which == 2) {
        // V: store transposed vT[b][n][s]
        #pragma unroll
        for (int nt = 0; nt < 8; ++nt) {
            float bb = bias[nt * 16 + lm];
            #pragma unroll
            for (int r = 0; r < 8; ++r) {
                int row  = m0 + r + 8 * hi;        // global row in [0,16384)
                int bidx = row >> 12;              // / 4096
                int s    = row & (S_ - 1);
                outb[((size_t)bidx * DH + nt * 16 + lm) * S_ + s] =
                    (__bf16)(acc[nt][r] + bb);
            }
        }
    } else {
        #pragma unroll
        for (int nt = 0; nt < 8; ++nt) {
            float bb = bias[nt * 16 + lm];
            #pragma unroll
            for (int r = 0; r < 8; ++r) {
                int row = m0 + r + 8 * hi;
                outb[(size_t)row * DH + nt * 16 + lm] =
                    (__bf16)((acc[nt][r] + bb) * sc);
            }
        }
    }
}

// ---------------------------------------------------------------------------
// Kernel 2: flash attention over bf16 q/k/vT in workspace.
// grid (S/128, B); each wave owns 16 query rows, loops 64-key tiles with
// double-buffered async (global->LDS) staging of K and V^T tiles.
// ---------------------------------------------------------------------------
__global__ __launch_bounds__(256)
void flash_attn_kernel(const unsigned short* __restrict__ ws_qkv,
                       float* __restrict__ out) {
    __shared__ __bf16 lds_k [2][KT * 128];   // K tiles, row-major     (2x16 KB)
    __shared__ __bf16 lds_vt[2][128 * KT];   // V^T tiles, d-major     (2x16 KB)
    __shared__ __bf16 lds_p [8 * 16 * KT];   // per-wave P scratch     (16 KB)

    const __bf16* qb = reinterpret_cast<const __bf16*>(ws_qkv);
    const __bf16* kb = qb + (size_t)MTOT * DH;
    const __bf16* vb = kb + (size_t)MTOT * DH;   // vT layout [B][DH][S]

    const int b    = blockIdx.y;
    const int tid  = threadIdx.x;
    const int wv   = tid >> 5;
    const int lane = tid & 31;
    const int hi   = lane >> 4;
    const int lm   = lane & 15;
    const int m0   = blockIdx.x * 128 + wv * 16;

    const __bf16* kbb = kb + (size_t)b * S_ * DH;
    const __bf16* vTb = vb + (size_t)b * DH * S_;

    // --- preload Q fragments (d = 4 steps of 32), resident for whole kernel ---
    const __bf16* qrow = qb + ((size_t)b * S_ + m0 + lm) * DH;
    v16bf Aq[4];
    #pragma unroll
    for (int ds = 0; ds < 4; ++ds) {
        v8bf r0 = *(const v8bf*)(qrow + ds * 32 + 8 * hi);
        v8bf r1 = *(const v8bf*)(qrow + ds * 32 + 16 + 8 * hi);
        Aq[ds] = mk16(r0, r1);
    }

    v8f Oacc[8] = {};
    float mrow[8], lrow[8];
    #pragma unroll
    for (int r = 0; r < 8; ++r) { mrow[r] = -1e30f; lrow[r] = 0.0f; }

    __bf16* pbuf = lds_p + wv * 16 * KT;

    // Issue async copies for (K, V^T) tile starting at key kt_ into buffer bufi.
    auto issue_stage = [&](int bufi, int kt_) {
        const __bf16* kt_g = kbb + (size_t)kt_ * DH;
        #pragma unroll
        for (int c = 0; c < 4; ++c) {            // K tile: 1024 x 16B chunks
            int i = tid + c * 256;
            async_copy_b128(lds_off(&lds_k[bufi][i * 8]), kt_g + i * 8);
        }
        #pragma unroll
        for (int c = 0; c < 4; ++c) {            // V^T tile: [128][KT], strided rows
            int i = tid + c * 256;               // chunk 0..1023
            int d = i >> 3, part = i & 7;
            async_copy_b128(lds_off(&lds_vt[bufi][d * KT + part * 8]),
                            vTb + (size_t)d * S_ + kt_ + part * 8);
        }
    };

    issue_stage(0, 0);

    for (int t = 0; t < S_ / KT; ++t) {
        const int cur = t & 1;
        wait_async0();          // this wave's copies for buffer `cur` are done
        __syncthreads();        // => everyone's copies landed; prev compute done
        if (t + 1 < S_ / KT) issue_stage(cur ^ 1, (t + 1) * KT);

        const __bf16* lkc = lds_k[cur];
        const __bf16* lvc = lds_vt[cur];

        // --- S = Q K^T : four 16x16 key sub-tiles, 4 d-steps each ---
        v8f Sv[4] = {};
        #pragma unroll
        for (int ds = 0; ds < 4; ++ds) {
            #pragma unroll
            for (int nt = 0; nt < 4; ++nt) {
                const v8bf* kr = (const v8bf*)(lkc + (size_t)(nt * 16 + lm) * DH
                                               + ds * 32 + 8 * hi);
                v16bf bf = mk16(kr[0], kr[2]);
                Sv[nt] = __builtin_amdgcn_wmma_f32_16x16x32_bf16(
                             false, Aq[ds], false, bf, (short)0, Sv[nt], false, false);
            }
        }

        // --- online softmax (per row = reg r within half-wave) ---
        #pragma unroll
        for (int r = 0; r < 8; ++r) {
            float s0 = Sv[0][r], s1 = Sv[1][r], s2 = Sv[2][r], s3 = Sv[3][r];
            float mx = fmaxf(fmaxf(s0, s1), fmaxf(s2, s3));
            #pragma unroll
            for (int off = 8; off >= 1; off >>= 1)
                mx = fmaxf(mx, __shfl_xor(mx, off, 32));
            float nm   = fmaxf(mrow[r], mx);
            float corr = __expf(mrow[r] - nm);
            float p0 = __expf(s0 - nm);
            float p1 = __expf(s1 - nm);
            float p2 = __expf(s2 - nm);
            float p3 = __expf(s3 - nm);
            Sv[0][r] = p0; Sv[1][r] = p1; Sv[2][r] = p2; Sv[3][r] = p3;
            float sm = (p0 + p1) + (p2 + p3);
            #pragma unroll
            for (int off = 8; off >= 1; off >>= 1)
                sm += __shfl_xor(sm, off, 32);
            lrow[r] = lrow[r] * corr + sm;
            mrow[r] = nm;
            #pragma unroll
            for (int nd = 0; nd < 8; ++nd) Oacc[nd][r] *= corr;
        }

        // --- P: C-layout -> A-layout via per-wave LDS round-trip ---
        #pragma unroll
        for (int r = 0; r < 8; ++r) {
            #pragma unroll
            for (int nt = 0; nt < 4; ++nt)
                pbuf[(r + 8 * hi) * KT + nt * 16 + lm] = (__bf16)Sv[nt][r];
        }
        asm volatile("s_wait_dscnt 0" ::: "memory");   // same-wave LDS RAW
        v16bf Ap[2];
        #pragma unroll
        for (int kh = 0; kh < 2; ++kh) {
            v8bf p0 = *(const v8bf*)(pbuf + lm * KT + kh * 32 + 8 * hi);
            v8bf p1 = *(const v8bf*)(pbuf + lm * KT + kh * 32 + 16 + 8 * hi);
            Ap[kh] = mk16(p0, p1);
        }

        // --- O += P V : 8 d sub-tiles from V^T tile, 2 key-halves each ---
        #pragma unroll
        for (int nd = 0; nd < 8; ++nd) {
            #pragma unroll
            for (int kh = 0; kh < 2; ++kh) {
                const v8bf* vr = (const v8bf*)(lvc + (size_t)(nd * 16 + lm) * KT
                                               + kh * 32 + 8 * hi);
                v16bf bv_ = mk16(vr[0], vr[2]);
                Oacc[nd] = __builtin_amdgcn_wmma_f32_16x16x32_bf16(
                               false, Ap[kh], false, bv_, (short)0, Oacc[nd],
                               false, false);
            }
        }
        __syncthreads();   // all reads of buffer `cur` done before it is refilled
    }

    // --- epilogue: O / l, f32 store ---
    float* outp = out + ((size_t)b * S_ + m0) * DH;
    #pragma unroll
    for (int r = 0; r < 8; ++r) {
        float inv = 1.0f / lrow[r];
        int row = r + 8 * hi;
        #pragma unroll
        for (int nd = 0; nd < 8; ++nd)
            outp[(size_t)row * DH + nd * 16 + lm] = Oacc[nd][r] * inv;
    }
}

// ---------------------------------------------------------------------------
extern "C" void kernel_launch(void* const* d_in, const int* in_sizes, int n_in,
                              void* d_out, int out_size, void* d_ws, size_t ws_size,
                              hipStream_t stream) {
    (void)in_sizes; (void)n_in; (void)out_size; (void)ws_size;
    const float* query = (const float*)d_in[0];
    const float* key_  = (const float*)d_in[1];
    const float* value = (const float*)d_in[2];
    const float* Wq    = (const float*)d_in[3];
    const float* bq    = (const float*)d_in[4];
    const float* Wk    = (const float*)d_in[5];
    const float* bk    = (const float*)d_in[6];
    const float* Wv    = (const float*)d_in[7];
    const float* bv    = (const float*)d_in[8];

    unsigned short* ws_qkv = (unsigned short*)d_ws;   // 3 * 16384*128 bf16 = 12.6 MB

    dim3 g1(MTOT / 128, 3), blk(256);
    qkv_proj_kernel<<<g1, blk, 0, stream>>>(query, key_, value,
                                            Wq, bq, Wk, bk, Wv, bv, ws_qkv);

    dim3 g2(S_ / 128, B_);
    flash_attn_kernel<<<g2, blk, 0, stream>>>(ws_qkv, (float*)d_out);
}
